// Timesfm2P5Attention_51221779972789
// MI455X (gfx1250) — matrix operand order repaired and checked by reading, not compiled, not run on hardware
//
#include <hip/hip_runtime.h>
#include <hip/hip_bf16.h>
#include <math.h>

#define BBATCH 8
#define SSEQ   1024
#define HIDN   1280
#define NH     16
#define HD     80
#define DP     96          // head dim padded to 3*32 for bf16 WMMA K-steps

typedef __bf16 bf16;
typedef __attribute__((ext_vector_type(16))) __bf16 v16bf;
typedef __attribute__((ext_vector_type(8)))  float  v8f;

// ---------------------------------------------------------------- WMMA helpers

static __device__ __forceinline__ v8f wmma_bf16(v16bf a, v16bf b, v8f c) {
  // D = A(16x32 bf16) * B(32x16 bf16) + C(16x16 f32)
  return __builtin_amdgcn_wmma_f32_16x16x32_bf16(false, a, false, b, (short)0, c,
                                                 false, false);
}

// A fragment (16x32, bf16), row-major source, leading dim ld (elements).
// Layout: lane M = lane%16; lanes0-15 hold K=0..15, lanes16-31 hold K=16..31
// (contiguous, matching frag_bt).
static __device__ __forceinline__ v16bf frag_a(const bf16* src, int ld, int row0, int k0) {
  const int lane = threadIdx.x & 31;
  const int m    = lane & 15;
  const int koff = (lane < 16) ? 0 : 16;
  const bf16* p  = src + (size_t)(row0 + m) * (size_t)ld + k0;
  v16bf a;
#pragma unroll
  for (int v = 0; v < 8; ++v) {
    const int kb = koff + 2 * v;
    a[2 * v]     = p[kb];
    a[2 * v + 1] = p[kb + 1];
  }
  return a;
}

// B fragment (32x16, bf16) where B[k][n] = srcT[n][k]; srcT row-major with ld.
// lane N = lane%16; lanes0-15 hold K=0..15, lanes16-31 hold K=16..31 (contiguous).
static __device__ __forceinline__ v16bf frag_bt(const bf16* srcT, int ld, int n0, int k0) {
  const int lane = threadIdx.x & 31;
  const int n    = lane & 15;
  const int base = (lane < 16) ? 0 : 16;
  const bf16* p  = srcT + (size_t)(n0 + n) * (size_t)ld + k0 + base;
  v16bf b;
#pragma unroll
  for (int i = 0; i < 16; ++i) b[i] = p[i];
  return b;
}

// ---------------------------------------------------------------- small kernels

__global__ void k_nummask(const int* __restrict__ mask, int* __restrict__ nm) {
  __shared__ int sdata[256];
  const int b = blockIdx.x, t = threadIdx.x;
  int sum = 0;
  for (int i = t; i < SSEQ; i += 256) sum += (mask[b * SSEQ + i] != 0) ? 1 : 0;
  sdata[t] = sum;
  __syncthreads();
  for (int s = 128; s > 0; s >>= 1) {
    if (t < s) sdata[t] += sdata[t + s];
    __syncthreads();
  }
  if (t == 0) nm[b] = sdata[0];
}

__global__ void k_f2bf(const float* __restrict__ in, bf16* __restrict__ out, int n) {
  int i = blockIdx.x * 256 + threadIdx.x;
  if (i < n) out[i] = (bf16)in[i];
}

// W (K=1280 x N=1280 row-major f32) -> Wt (N x K row-major bf16)
__global__ void k_wtrans(const float* __restrict__ in, bf16* __restrict__ out) {
  int i = blockIdx.x * 256 + threadIdx.x;
  if (i < HIDN * HIDN) {
    int k = i / HIDN, n = i % HIDN;
    out[(size_t)n * HIDN + k] = (bf16)in[i];
  }
}

// ---------------------------------------------------------------- QKV GEMM + epilogue
// One wave computes a 32-row x 80-col (one head) tile of X@W, then applies
// RMSNorm (+q scaling) + RoPE and writes padded bf16 tensors for attention.
// mode 0 = Q -> Qp(b,h,s,96)   1 = K -> Kp(b,h,s,96)   2 = V -> Vt(b,h,96,s)

__global__ void k_qkv(const bf16* __restrict__ Xb,
                      const bf16* __restrict__ Wqt, const bf16* __restrict__ Wkt,
                      const bf16* __restrict__ Wvt,
                      const float* __restrict__ qns, const float* __restrict__ kns,
                      const float* __restrict__ scaling, const int* __restrict__ nm,
                      bf16* __restrict__ Qp, bf16* __restrict__ Kp,
                      bf16* __restrict__ Vt) {
  const int mode = blockIdx.z;
  const int h    = blockIdx.y;
  const int row0 = blockIdx.x * 32;
  const int lane = threadIdx.x & 31;
  const bf16* Wt = (mode == 0) ? Wqt : (mode == 1) ? Wkt : Wvt;

  v8f c[2][5];
#pragma unroll
  for (int mt = 0; mt < 2; ++mt)
#pragma unroll
    for (int nt = 0; nt < 5; ++nt) c[mt][nt] = (v8f){};

#pragma unroll 1
  for (int kk = 0; kk < HIDN; kk += 32) {
    __builtin_prefetch(Xb + (size_t)(row0 + lane) * HIDN + kk + 64, 0, 0);
    v16bf a0 = frag_a(Xb, HIDN, row0, kk);
    v16bf a1 = frag_a(Xb, HIDN, row0 + 16, kk);
#pragma unroll
    for (int nt = 0; nt < 5; ++nt) {
      v16bf bf = frag_bt(Wt, HIDN, h * HD + nt * 16, kk);
      c[0][nt] = wmma_bf16(a0, bf, c[0][nt]);
      c[1][nt] = wmma_bf16(a1, bf, c[1][nt]);
    }
  }

  __shared__ float tile[32][84];
  const int hi8 = (lane < 16) ? 0 : 8;
#pragma unroll
  for (int mt = 0; mt < 2; ++mt)
#pragma unroll
    for (int nt = 0; nt < 5; ++nt)
#pragma unroll
      for (int r = 0; r < 8; ++r)
        tile[mt * 16 + r + hi8][nt * 16 + (lane & 15)] = c[mt][nt][r];
  __syncthreads();

  // epilogue: lane owns one row
  const int gidx = row0 + lane;
  const int bb = gidx / SSEQ, ss = gidx % SSEQ;
  float x[HD];
#pragma unroll
  for (int d = 0; d < HD; ++d) x[d] = tile[lane][d];

  if (mode < 2) {
    float ms = 0.f;
#pragma unroll
    for (int d = 0; d < HD; ++d) ms += x[d] * x[d];
    const float inv = rsqrtf(ms * (1.0f / HD) + 1e-6f);
    const float* nsc = (mode == 0) ? qns : kns;
#pragma unroll
    for (int d = 0; d < HD; ++d) x[d] *= inv * nsc[d];
    if (mode == 0) {
      const float qk = 1.442695041f / sqrtf((float)HD);
#pragma unroll
      for (int d = 0; d < HD; ++d) {
        const float sv = scaling[d];
        const float sp = (sv > 20.f) ? sv : log1pf(expf(sv));
        x[d] *= sp * qk;
      }
    }
    const float pos = (float)(ss - nm[bb]);
    bf16* dst = ((mode == 0) ? Qp : Kp) +
                ((size_t)(bb * NH + h) * SSEQ + ss) * (size_t)DP;
#pragma unroll
    for (int j = 0; j < HD / 2; ++j) {
      const float its = exp2f(-(float)j * (13.28771238f / 40.0f)); // 1/10000^(j/40)
      const float arg = pos * its;
      const float sn = sinf(arg), cs = cosf(arg);
      const float x1 = x[j], x2 = x[j + HD / 2];
      dst[j]          = (bf16)(x1 * cs - x2 * sn);
      dst[j + HD / 2] = (bf16)(x2 * cs + x1 * sn);
    }
#pragma unroll
    for (int d = HD; d < DP; ++d) dst[d] = (bf16)0.0f;
  } else {
    bf16* dst = Vt + (size_t)(bb * NH + h) * DP * SSEQ;
#pragma unroll
    for (int d = 0; d < DP; ++d)
      dst[(size_t)d * SSEQ + ss] = (d < HD) ? (bf16)x[d] : (bf16)0.0f;
  }
}

// ---------------------------------------------------------------- attention core
// One wave per (b, h, 32 query rows). Flash-style online softmax via LDS.

__global__ void k_attn(const bf16* __restrict__ Qp, const bf16* __restrict__ Kp,
                       const bf16* __restrict__ Vt, const int* __restrict__ nm,
                       bf16* __restrict__ Ob) {
  const int lane = threadIdx.x & 31;
  const int q0 = blockIdx.x * 32;
  const int h  = blockIdx.y;
  const int b  = blockIdx.z;
  const bf16* Qh = Qp + (size_t)(b * NH + h) * SSEQ * DP;
  const bf16* Kh = Kp + (size_t)(b * NH + h) * SSEQ * DP;
  const bf16* Vh = Vt + (size_t)(b * NH + h) * DP * SSEQ;
  const int nmask = nm[b];
  const int hi8 = (lane < 16) ? 0 : 8;

  // Q A-fragments live in registers for the whole kv loop
  v16bf qa[2][3];
#pragma unroll
  for (int mt = 0; mt < 2; ++mt)
#pragma unroll
    for (int dk = 0; dk < 3; ++dk) qa[mt][dk] = frag_a(Qh, DP, q0 + mt * 16, dk * 32);

  __shared__ float sS[32][33];
  __shared__ bf16  sP[32][32];
  __shared__ float sScale[32];
  __shared__ float sInv[32];

  float rm = -1e30f, rl = 0.0f;   // lane owns row = lane
  v8f o[2][5];
#pragma unroll
  for (int mt = 0; mt < 2; ++mt)
#pragma unroll
    for (int nt = 0; nt < 5; ++nt) o[mt][nt] = (v8f){};

#pragma unroll 1
  for (int kv0 = 0; kv0 < q0 + 32; kv0 += 32) {
    // ---- scores = Q * K^T (padded d contributes zeros)
    v8f sc[2][2];
#pragma unroll
    for (int mt = 0; mt < 2; ++mt)
#pragma unroll
      for (int nk = 0; nk < 2; ++nk) sc[mt][nk] = (v8f){};
#pragma unroll
    for (int dk = 0; dk < 3; ++dk) {
      v16bf kb0 = frag_bt(Kh, DP, kv0, dk * 32);
      v16bf kb1 = frag_bt(Kh, DP, kv0 + 16, dk * 32);
#pragma unroll
      for (int mt = 0; mt < 2; ++mt) {
        sc[mt][0] = wmma_bf16(qa[mt][dk], kb0, sc[mt][0]);
        sc[mt][1] = wmma_bf16(qa[mt][dk], kb1, sc[mt][1]);
      }
    }
    // ---- masked scores -> LDS
#pragma unroll
    for (int mt = 0; mt < 2; ++mt)
#pragma unroll
      for (int nk = 0; nk < 2; ++nk)
#pragma unroll
        for (int r = 0; r < 8; ++r) {
          const int row = mt * 16 + r + hi8;
          const int col = nk * 16 + (lane & 15);
          const int qi = q0 + row, ki = kv0 + col;
          sS[row][col] = (qi >= ki && ki >= nmask) ? sc[mt][nk][r] : -1e30f;
        }
    __syncthreads();
    // ---- online softmax: lane handles row = lane
    float mx = rm;
#pragma unroll
    for (int j = 0; j < 32; ++j) mx = fmaxf(mx, sS[lane][j]);
    const float fac = expf(rm - mx);
    float sum = 0.f;
#pragma unroll
    for (int j = 0; j < 32; ++j) {
      const float sv = sS[lane][j];
      const float p = (sv > -1e29f) ? expf(sv - mx) : 0.0f;
      sP[lane][j] = (bf16)p;
      sum += p;
    }
    rl = rl * fac + sum;
    rm = mx;
    sScale[lane] = fac;
    __syncthreads();
    // ---- rescale accumulators, then o += P * V
#pragma unroll
    for (int mt = 0; mt < 2; ++mt)
#pragma unroll
      for (int nt = 0; nt < 5; ++nt)
#pragma unroll
        for (int r = 0; r < 8; ++r) o[mt][nt][r] *= sScale[mt * 16 + r + hi8];
    v16bf pa0 = frag_a(&sP[0][0], 32, 0, 0);
    v16bf pa1 = frag_a(&sP[0][0], 32, 16, 0);
#pragma unroll
    for (int nt = 0; nt < 5; ++nt) {
      v16bf vb = frag_bt(Vh, SSEQ, nt * 16, kv0);   // B[kv][d] = Vt[d][kv0+kv]
      o[0][nt] = wmma_bf16(pa0, vb, o[0][nt]);
      o[1][nt] = wmma_bf16(pa1, vb, o[1][nt]);
    }
    __syncthreads();
  }

  sInv[lane] = (rl > 0.f) ? (1.0f / rl) : 0.0f;
  __syncthreads();
#pragma unroll
  for (int mt = 0; mt < 2; ++mt)
#pragma unroll
    for (int nt = 0; nt < 5; ++nt)
#pragma unroll
      for (int r = 0; r < 8; ++r) {
        const int row = mt * 16 + r + hi8;
        const int col = nt * 16 + (lane & 15);
        const int s = q0 + row;
        Ob[((size_t)(b * SSEQ + s)) * HIDN + h * HD + col] =
            (bf16)(o[mt][nt][r] * sInv[row]);
      }
}

// ---------------------------------------------------------------- output projection
// 64x64 tile per wave: d_out = Ob(8192x1280 bf16) @ Wo -> f32

__global__ void k_oproj(const bf16* __restrict__ Ob, const bf16* __restrict__ Wot,
                        float* __restrict__ out) {
  const int lane = threadIdx.x & 31;
  const int m0 = blockIdx.x * 64;
  const int n0 = blockIdx.y * 64;
  v8f c[4][4];
#pragma unroll
  for (int mt = 0; mt < 4; ++mt)
#pragma unroll
    for (int nt = 0; nt < 4; ++nt) c[mt][nt] = (v8f){};

#pragma unroll 1
  for (int kk = 0; kk < HIDN; kk += 32) {
    __builtin_prefetch(Ob + (size_t)(m0 + lane) * HIDN + kk + 64, 0, 0);
    v16bf a[4], bfr[4];
#pragma unroll
    for (int mt = 0; mt < 4; ++mt) a[mt] = frag_a(Ob, HIDN, m0 + mt * 16, kk);
#pragma unroll
    for (int nt = 0; nt < 4; ++nt) bfr[nt] = frag_bt(Wot, HIDN, n0 + nt * 16, kk);
#pragma unroll
    for (int mt = 0; mt < 4; ++mt)
#pragma unroll
      for (int nt = 0; nt < 4; ++nt) c[mt][nt] = wmma_bf16(a[mt], bfr[nt], c[mt][nt]);
  }

  const int hi8 = (lane < 16) ? 0 : 8;
#pragma unroll
  for (int mt = 0; mt < 4; ++mt)
#pragma unroll
    for (int nt = 0; nt < 4; ++nt)
#pragma unroll
      for (int r = 0; r < 8; ++r) {
        const int row = m0 + mt * 16 + r + hi8;
        const int col = n0 + nt * 16 + (lane & 15);
        out[(size_t)row * HIDN + col] = c[mt][nt][r];
      }
}

// ---------------------------------------------------------------- host launcher

extern "C" void kernel_launch(void* const* d_in, const int* in_sizes, int n_in,
                              void* d_out, int out_size, void* d_ws, size_t ws_size,
                              hipStream_t stream) {
  const float* X    = (const float*)d_in[0];
  const float* scal = (const float*)d_in[1];
  const float* wq   = (const float*)d_in[2];
  const float* wk   = (const float*)d_in[3];
  const float* wv   = (const float*)d_in[4];
  const float* wo   = (const float*)d_in[5];
  const float* qns  = (const float*)d_in[6];
  const float* kns  = (const float*)d_in[7];
  const int*   pm   = (const int*)d_in[8];
  float* outp = (float*)d_out;

  char* ws = (char*)d_ws;
  const size_t sz_xb = (size_t)BBATCH * SSEQ * HIDN * sizeof(bf16);
  const size_t sz_w  = (size_t)HIDN * HIDN * sizeof(bf16);
  const size_t sz_qk = (size_t)BBATCH * NH * SSEQ * DP * sizeof(bf16);
  size_t off = 0;
  int* nm = (int*)(ws + off);          off += 256;
  bf16* Xb  = (bf16*)(ws + off);       off += sz_xb;
  bf16* Wqt = (bf16*)(ws + off);       off += sz_w;
  bf16* Wkt = (bf16*)(ws + off);       off += sz_w;
  bf16* Wvt = (bf16*)(ws + off);       off += sz_w;
  bf16* Wot = (bf16*)(ws + off);       off += sz_w;
  bf16* Qp  = (bf16*)(ws + off);       off += sz_qk;
  bf16* Kp  = (bf16*)(ws + off);       off += sz_qk;
  bf16* Vt  = (bf16*)(ws + off);       off += sz_qk;
  bf16* Ob  = (bf16*)(ws + off);       off += sz_xb;
  (void)ws_size; (void)n_in; (void)in_sizes; (void)out_size;

  k_nummask<<<BBATCH, 256, 0, stream>>>(pm, nm);

  const int nx = BBATCH * SSEQ * HIDN;
  k_f2bf<<<(nx + 255) / 256, 256, 0, stream>>>(X, Xb, nx);
  const int nw = HIDN * HIDN;
  k_wtrans<<<(nw + 255) / 256, 256, 0, stream>>>(wq, Wqt);
  k_wtrans<<<(nw + 255) / 256, 256, 0, stream>>>(wk, Wkt);
  k_wtrans<<<(nw + 255) / 256, 256, 0, stream>>>(wv, Wvt);
  k_wtrans<<<(nw + 255) / 256, 256, 0, stream>>>(wo, Wot);

  k_qkv<<<dim3((BBATCH * SSEQ) / 32, NH, 3), 32, 0, stream>>>(
      Xb, Wqt, Wkt, Wvt, qns, kns, scal, nm, Qp, Kp, Vt);

  k_attn<<<dim3(SSEQ / 32, NH, BBATCH), 32, 0, stream>>>(Qp, Kp, Vt, nm, Ob);

  k_oproj<<<dim3((BBATCH * SSEQ) / 64, HIDN / 64), 32, 0, stream>>>(Ob, Wot, outp);
}
